// QConv2d_32315333935634
// MI455X (gfx1250) — matrix operand order repaired and checked by reading, not compiled, not run on hardware
//
#include <hip/hip_runtime.h>
#include <cstdint>
#include <cmath>

typedef __attribute__((ext_vector_type(16))) _Float16 v16h;
typedef __attribute__((ext_vector_type(8)))  _Float16 v8h;
typedef __attribute__((ext_vector_type(8)))  float    v8f;

// ---------------- problem constants ----------------
#define N_IMG 8
#define C_IN 64
#define HW 32
#define OUT_CH 128
#define LCOLS 576           // C_IN*3*3
#define LPAD 640            // padded to 5*128
#define NSUB 5
#define SUB 128
#define ROWS 8192           // N*HW*HW
#define NABITS 7
#define NCELLS 3
#define ADC_BITS 5
#define LOWER_G 0.1f        // 1/ONOFF
#define VARI 0.05f
#define SAF_P 0.001f

// ---------------- helpers ----------------
__device__ __forceinline__ uint32_t hashu(uint32_t x) {
    x ^= x >> 16; x *= 0x7feb352du;
    x ^= x >> 15; x *= 0x846ca68bu;
    x ^= x >> 16; return x;
}
__device__ __forceinline__ float u01(uint32_t h) {
    return (float)(h >> 8) * (1.0f / 16777216.0f);
}
__device__ __forceinline__ float gauss01(uint32_t seed) {
    uint32_t h1 = hashu(seed ^ 0x68bc21ebu);
    uint32_t h2 = hashu(seed ^ 0x02e5be93u);
    float u1 = ((float)(h1 >> 8) + 1.0f) * (1.0f / 16777216.0f); // (0,1]
    float u2 = (float)(h2 >> 8) * (1.0f / 16777216.0f);
    return sqrtf(-2.0f * logf(u1)) * cosf(6.28318530718f * u2);
}
// order-preserving float<->uint keys (for signed-value atomic min/max)
__device__ __forceinline__ uint32_t fkey(float f) {
    uint32_t u = __float_as_uint(f);
    return (u & 0x80000000u) ? ~u : (u | 0x80000000u);
}
__device__ __forceinline__ float funkey(uint32_t k) {
    uint32_t u = (k & 0x80000000u) ? (k ^ 0x80000000u) : ~k;
    return __uint_as_float(u);
}

// ---------------- kernel 0: zero accumulator + max slot ----------------
__global__ void k_init(float* __restrict__ acc, unsigned* __restrict__ slots) {
    int t = blockIdx.x * 256 + threadIdx.x;          // 1048576 threads exact
    acc[t] = 0.0f;
    if (t == 0) slots[0] = 0u;                        // max(x) bits (x>=0)
}

// ---------------- kernel 1: global max of x (x >= 0) ----------------
__global__ void k_maxx(const float* __restrict__ x, unsigned* __restrict__ slots) {
    __shared__ unsigned sm;
    int t = blockIdx.x * 256 + threadIdx.x;           // 131072 threads, 4 elems each
    float m = 0.0f;
#pragma unroll
    for (int i = 0; i < 4; ++i) m = fmaxf(m, x[t + i * 131072]);
    if (threadIdx.x == 0) sm = 0u;
    __syncthreads();
    atomicMax(&sm, __float_as_uint(m));               // nonneg floats: bits ordered
    __syncthreads();
    if (threadIdx.x == 0) atomicMax(&slots[0], sm);
}

// ---------------- kernel 2: im2col + activation quantize (u8) ----------------
__global__ void k_im2col(const float* __restrict__ x, uint8_t* __restrict__ inputQ,
                         const unsigned* __restrict__ slots) {
    int t = blockIdx.x * 256 + threadIdx.x;           // ROWS*LPAD = 5242880 exact
    int row = t / LPAD;
    int col = t - row * LPAD;
    float a_scale = fmaxf(__uint_as_float(slots[0]), 1e-8f) / 127.0f;
    uint8_t q = 0;
    if (col < LCOLS) {
        int c = col / 9, r9 = col - c * 9;
        int ki = r9 / 3, kj = r9 - ki * 3;
        int n  = row >> 10, hw = row & 1023;
        int ho = hw >> 5,  wo = hw & 31;
        int h = ho - 1 + ki, w = wo - 1 + kj;
        if (h >= 0 && h < HW && w >= 0 && w < HW) {
            float v = x[(size_t)(n * C_IN + c) * 1024 + h * HW + w];
            q = (uint8_t)rintf(v / a_scale);
        }
    }
    inputQ[t] = q;
}

// ---------------- kernel 3: weight quantize per (out, subarray) ----------------
__global__ void k_wquant(const float* __restrict__ weight, int8_t* __restrict__ wq,
                         float* __restrict__ w_scale) {
    int t = blockIdx.x * 256 + threadIdx.x;
    if (t >= OUT_CH * NSUB) return;
    int out = t / NSUB, s = t - out * NSUB;
    float amax = 0.0f;
    for (int cc = 0; cc < SUB; ++cc) {
        int col = s * SUB + cc;
        if (col < LCOLS) amax = fmaxf(amax, fabsf(weight[out * LCOLS + col]));
    }
    float ws = fmaxf(amax, 1e-8f) / 127.0f;
    w_scale[t] = ws;
    for (int cc = 0; cc < SUB; ++cc) {
        int col = s * SUB + cc;
        int8_t q = 0;
        if (col < LCOLS) q = (int8_t)rintf(weight[out * LCOLS + col] / ws);
        wq[out * LPAD + col] = q;
    }
}

// ---------------- kernel 4: precompute rQ f16 planes for all 21 (z,k) ----------------
__global__ void k_rqgen(const int8_t* __restrict__ wq, _Float16* __restrict__ rQ) {
    int t = blockIdx.x * 256 + threadIdx.x;           // OUT_CH*LPAD = 81920 exact
    int q = (int)wq[t];
    int rem[NCELLS];
    bool sa0[NCELLS], sa1[NCELLS];
#pragma unroll
    for (int k = 0; k < NCELLS; ++k) {
        rem[k] = q % 4;                                // trunc like jnp.fmod
        q = q / 4;
        float u = u01(hashu(0x5AF00000u + (uint32_t)(k * 81920 + t)));
        sa0[k] = (u < SAF_P);
        sa1[k] = (u > 1.0f - SAF_P);
    }
    for (int z = 0; z < NABITS; ++z) {
#pragma unroll
        for (int k = 0; k < NCELLS; ++k) {
            uint32_t seed = (uint32_t)((z * NCELLS + k) * 81920 + t) ^ 0xA5A5A5A5u;
            float v = (float)rem[k] * (1.0f + VARI * gauss01(seed));
            if (sa0[k]) v = LOWER_G;
            else if (sa1[k]) v = (v > 0.0f) ? 1.0f : ((v < 0.0f) ? -1.0f : 0.0f);
            rQ[(size_t)(z * NCELLS + k) * (OUT_CH * LPAD) + t] = (_Float16)v;
        }
    }
}

// ---------------- kernel 5: reset ADC min/max keys ----------------
__global__ void k_reset_mnmx(unsigned* __restrict__ mnmx) {
    mnmx[0] = 0xFF7FFFFFu;   // fkey(+FLT_MAX)
    mnmx[1] = 0x00800000u;   // fkey(-FLT_MAX)
}

// ---------------- kernel 6: WMMA partial matmul + global min/max ----------------
// grid (64 row-tiles, 5 subarrays), 256 threads (8 waves of 32)
__global__ void k_wmma_partial(const uint8_t* __restrict__ inputQ,
                               const _Float16* __restrict__ rQzk,
                               float* __restrict__ partial,
                               unsigned* __restrict__ mnmx, int z) {
    __shared__ alignas(32) _Float16 As[128][32];   // [row_in_tile][k]  bits as f16
    __shared__ alignas(32) _Float16 Bs[128][32];   // [out][k]          rQ tile
    __shared__ unsigned s_mn, s_mx;

    const int s = blockIdx.y;
    const int rowbase = blockIdx.x * 128;
    const int tid = threadIdx.x;
    const int wave = tid >> 5, lane = tid & 31;
    const int colbase = s * SUB;

    v8f acc[8];
    const v8f vzero = {0.f, 0.f, 0.f, 0.f, 0.f, 0.f, 0.f, 0.f};
#pragma unroll
    for (int i = 0; i < 8; ++i) acc[i] = vzero;

    for (int kb = 0; kb < SUB; kb += 32) {
        // cooperative staging: 256 threads, each moves 16 elements
        {
            int r = tid >> 1, part = tid & 1;
            // A: u8 quantized activations -> bit z -> f16 {0,1}
            const uint8_t* gp = inputQ + (size_t)(rowbase + r) * LPAD + colbase + kb + part * 16;
            uint4 bytes = *(const uint4*)gp;
            unsigned bw[4] = {bytes.x, bytes.y, bytes.z, bytes.w};
            v16h av;
#pragma unroll
            for (int i = 0; i < 16; ++i) {
                unsigned byte = (bw[i >> 2] >> ((i & 3) * 8)) & 0xFFu;
                av[i] = (_Float16)(float)((byte >> z) & 1u);
            }
            *(v16h*)&As[r][part * 16] = av;
            // B: f16 cell weights, tile copy (r doubles as out channel 0..127)
            const _Float16* gq = rQzk + (size_t)r * LPAD + colbase + kb + part * 16;
            *(v16h*)&Bs[r][part * 16] = *(const v16h*)gq;
        }
        __syncthreads();

        // A fragment: lanes 0-15 -> K 0..15; lanes 16-31 -> K 16..31
        const int mrow = wave * 16 + (lane & 15);
        const int akb = (lane < 16) ? 0 : 16;
        v8h alo = *(const v8h*)&As[mrow][akb];
        v8h ahi = *(const v8h*)&As[mrow][akb + 8];
        v16h afrag = __builtin_shufflevector(alo, ahi,
                        0, 1, 2, 3, 4, 5, 6, 7, 8, 9, 10, 11, 12, 13, 14, 15);
        // B fragment: lane -> N = lane&15; half-wave selects K 0..15 / 16..31
        const int bkb = (lane < 16) ? 0 : 16;
#pragma unroll
        for (int ot = 0; ot < 8; ++ot) {
            const int n = ot * 16 + (lane & 15);
            v16h bfrag = *(const v16h*)&Bs[n][bkb];
            acc[ot] = __builtin_amdgcn_wmma_f32_16x16x32_f16(
                false, afrag, false, bfrag, (short)0, acc[ot], false, false);
        }
        __syncthreads();
    }

    // epilogue: store partial, reduce min/max
    float lmn = 3.4e38f, lmx = -3.4e38f;
    const int mbase = rowbase + wave * 16 + ((lane >> 4) << 3);
    const int ncol = lane & 15;
#pragma unroll
    for (int ot = 0; ot < 8; ++ot) {
        const int out = ot * 16 + ncol;
#pragma unroll
        for (int r = 0; r < 8; ++r) {
            float v = acc[ot][r];
            lmn = fminf(lmn, v);
            lmx = fmaxf(lmx, v);
            partial[((size_t)s * ROWS + (mbase + r)) * OUT_CH + out] = v;
        }
    }
    if (tid == 0) { s_mn = 0xFFFFFFFFu; s_mx = 0u; }
    __syncthreads();
    atomicMin(&s_mn, fkey(lmn));
    atomicMax(&s_mx, fkey(lmx));
    __syncthreads();
    if (tid == 0) {
        atomicMin(&mnmx[0], s_mn);
        atomicMax(&mnmx[1], s_mx);
    }
}

// ---------------- kernel 7: ADC quantize, per-subarray rescale, accumulate ----------------
__global__ void k_quant_acc(const float* __restrict__ partial,
                            const unsigned* __restrict__ mnmx,
                            const float* __restrict__ w_scale,
                            float* __restrict__ acc, float kz) {
    int t = blockIdx.x * 256 + threadIdx.x;           // ROWS*OUT_CH exact
    int row = t >> 7, out = t & 127;
    float mn = funkey(mnmx[0]);
    float mx = funkey(mnmx[1]);
    float step = (mx - mn) * 0.03125f;                // * 2^-ADC_BITS
    if (step <= 0.0f) step = 1.0f;
    float rstep = 1.0f / step;
    float sum = 0.0f;
#pragma unroll
    for (int s = 0; s < NSUB; ++s) {
        float v = partial[((size_t)s * ROWS + row) * OUT_CH + out];
        float idx = floorf((v - mn) * rstep);
        idx = fminf(fmaxf(idx, 0.0f), 31.0f);
        sum += (idx * step + mn) * w_scale[out * NSUB + s];
    }
    acc[t] += sum * kz;
}

// ---------------- kernel 8: bias + a_scale + NCHW transpose ----------------
__global__ void k_final(const float* __restrict__ acc, const float* __restrict__ bias,
                        const unsigned* __restrict__ slots, float* __restrict__ out) {
    int t = blockIdx.x * 256 + threadIdx.x;           // 1048576 exact, output-linear
    int w = t & 31, h = (t >> 5) & 31, o = (t >> 10) & 127, n = t >> 17;
    float a_scale = fmaxf(__uint_as_float(slots[0]), 1e-8f) / 127.0f;
    int row = n * 1024 + h * 32 + w;
    out[t] = acc[row * OUT_CH + o] * a_scale + bias[o];
}

// ---------------- host side ----------------
extern "C" void kernel_launch(void* const* d_in, const int* in_sizes, int n_in,
                              void* d_out, int out_size, void* d_ws, size_t ws_size,
                              hipStream_t stream) {
    const float* x    = (const float*)d_in[0];
    const float* wgt  = (const float*)d_in[1];
    const float* bias = (const float*)d_in[2];
    float* out = (float*)d_out;
    char* ws = (char*)d_ws;

    // workspace layout (256B-aligned regions, ~34 MB total)
    const size_t OFF_INPUTQ  = 256;
    const size_t OFF_WQ      = OFF_INPUTQ + (size_t)ROWS * LPAD;            // 5 242 880
    const size_t OFF_WSCALE  = OFF_WQ + (size_t)OUT_CH * LPAD;              // 81 920
    const size_t OFF_RQ      = OFF_WSCALE + 2560;
    const size_t OFF_PARTIAL = OFF_RQ + (size_t)NABITS * NCELLS * OUT_CH * LPAD * 2;
    const size_t OFF_ACC     = OFF_PARTIAL + (size_t)NSUB * ROWS * OUT_CH * 4;

    unsigned* slots    = (unsigned*)ws;          // [0]=max(x) bits, [2..3]=ADC min/max keys
    unsigned* mnmx     = slots + 2;
    uint8_t*  inputQ   = (uint8_t*)(ws + OFF_INPUTQ);
    int8_t*   wq       = (int8_t*)(ws + OFF_WQ);
    float*    w_scale  = (float*)(ws + OFF_WSCALE);
    _Float16* rQall    = (_Float16*)(ws + OFF_RQ);
    float*    partial  = (float*)(ws + OFF_PARTIAL);
    float*    acc      = (float*)(ws + OFF_ACC);

    k_init  <<<4096, 256, 0, stream>>>(acc, slots);
    k_maxx  <<<512, 256, 0, stream>>>(x, slots);
    k_im2col<<<(ROWS * LPAD) / 256, 256, 0, stream>>>(x, inputQ, slots);
    k_wquant<<<3, 256, 0, stream>>>(wgt, wq, w_scale);
    k_rqgen <<<(OUT_CH * LPAD) / 256, 256, 0, stream>>>(wq, rQall);

    for (int z = 0; z < NABITS; ++z) {
        for (int k = 0; k < NCELLS; ++k) {
            const int zk = z * NCELLS + k;
            const float kz = (float)(1u << (2 * k)) * (float)(1u << z); // 4^k * 2^z
            k_reset_mnmx<<<1, 1, 0, stream>>>(mnmx);
            k_wmma_partial<<<dim3(ROWS / 128, NSUB), 256, 0, stream>>>(
                inputQ, rQall + (size_t)zk * OUT_CH * LPAD, partial, mnmx, z);
            k_quant_acc<<<(ROWS * OUT_CH) / 256, 256, 0, stream>>>(
                partial, mnmx, w_scale, acc, kz);
        }
    }
    k_final<<<4096, 256, 0, stream>>>(acc, bias, slots, out);
}
